// NNConvModel_52578989638121
// MI455X (gfx1250) — matrix operand order repaired and checked by reading, not compiled, not run on hardware
//
#include <hip/hip_runtime.h>

// ---------------- problem constants ----------------
constexpr int B    = 1024;
constexpr int C    = 32;
constexpr int G    = 32;
constexpr int CONV = 32;
constexpr int NT   = 30;
constexpr int OH   = 30, OW = 30;
constexpr int F    = CONV * OH * OW + NT;   // 28830
constexpr int KP   = 28928;                 // K padded to multiple of 128 (226 stages)
constexpr int T    = 16;
constexpr int A    = 10;
constexpr int HA   = 128;
constexpr int HC   = 32;
constexpr int NOUT = HA + T * HC;           // 640 fused output columns
constexpr int NS   = KP / 128;              // 226 K-stages
constexpr int LDSW = 136;                   // LDS row stride (bf16) = 128 + 8 pad

// ---------------- CDNA5 async-to-LDS availability ----------------
#if defined(__has_builtin)
#  if __has_builtin(__builtin_amdgcn_global_load_async_to_lds_b128) && \
      __has_builtin(__builtin_amdgcn_s_wait_asynccnt)
#    define USE_ASYNC 1
#  endif
#endif
#ifndef USE_ASYNC
#  define USE_ASYNC 0
#endif

// ---------------- vector types ----------------
typedef __attribute__((ext_vector_type(16))) __bf16 v16bf;
typedef __attribute__((ext_vector_type(8)))  __bf16 v8bf;
typedef __attribute__((ext_vector_type(8)))  float  v8f;
typedef __attribute__((ext_vector_type(4)))  int    v4i;

__device__ inline unsigned short f2bf(float x) {
    union { float f; unsigned u; } v; v.f = x;
    unsigned r = v.u + 0x7FFFu + ((v.u >> 16) & 1u);   // round-to-nearest-even
    return (unsigned short)(r >> 16);
}

// ------- fused weight conversion: [640][KP] bf16, rows 0..127 = actor_w1^T,
// ------- rows 128+t*32+n = critic_w1[t,:,n]; K zero-padded to KP -------
__global__ void cvt_weights(const float* __restrict__ aw1,
                            const float* __restrict__ cw1,
                            unsigned short* __restrict__ out) {
    size_t i = (size_t)blockIdx.x * 256 + threadIdx.x;
    if (i >= (size_t)NOUT * KP) return;
    int row = (int)(i / KP), k = (int)(i % KP);
    float v = 0.0f;
    if (k < F) {
        if (row < HA) {
            v = aw1[(size_t)k * HA + row];
        } else {
            int t = (row - HA) >> 5, n = (row - HA) & 31;
            v = cw1[((size_t)t * F + k) * HC + n];
        }
    }
    out[i] = f2bf(v);
}

// ---------------- conv3x3 VALID + relu + featurize -> bf16 feats [B][KP] ----------------
__global__ __launch_bounds__(128)
void conv_feat(const float* __restrict__ x, const float* __restrict__ cw,
               const float* __restrict__ cb, const float* __restrict__ flat,
               unsigned short* __restrict__ feats) {
    __shared__ float w[C * 9];
    const int b = blockIdx.x, co = blockIdx.y;
    const int tid = threadIdx.x;
    for (int i = tid; i < C * 9; i += 128) w[i] = cw[co * C * 9 + i];
    __syncthreads();
    const float bias = cb[co];
    const float* xb = x + (size_t)b * C * G * G;
    unsigned short* fb = feats + (size_t)b * KP;
    for (int p = tid; p < OH * OW; p += 128) {
        int y = p / OW, xx = p % OW;
        float acc = bias;
        #pragma unroll 4
        for (int ci = 0; ci < C; ++ci) {
            const float* xc = xb + ci * G * G + y * G + xx;
            const float* wc = w + ci * 9;
            acc += xc[0] * wc[0] + xc[1] * wc[1] + xc[2] * wc[2]
                 + xc[G + 0] * wc[3] + xc[G + 1] * wc[4] + xc[G + 2] * wc[5]
                 + xc[2 * G + 0] * wc[6] + xc[2 * G + 1] * wc[7] + xc[2 * G + 2] * wc[8];
        }
        acc = fmaxf(acc, 0.0f);
        fb[co * OH * OW + p] = f2bf(acc);
    }
    if (co == 0) {   // key-feature tail (30 values) + zero padding up to KP
        float v = (tid < NT) ? flat[b * NT + tid] : 0.0f;
        fb[CONV * OH * OW + tid] = f2bf(v);   // covers 28800 .. 28927
    }
}

// ---------------- WMMA fragment loads (CDNA5 16-bit layouts, ISA 7.12.2) ----------------
// A 16x32: lane l holds row m=l%16; K = {base..base+7} in VGPR0-3, {16+base..16+base+7}
// in VGPR4-7 where base = (l>=16)*8.
__device__ inline v16bf load_a_frag(const unsigned short* __restrict__ row, int k0, int lane) {
    const int base = (lane & 16) ? 8 : 0;
    v8bf lo = *(const v8bf*)(row + k0 + base);
    v8bf hi = *(const v8bf*)(row + k0 + 16 + base);
    v16bf a;
    #pragma unroll
    for (int i = 0; i < 8; ++i) { a[i] = lo[i]; a[8 + i] = hi[i]; }
    return a;
}
// B 32x16 fragment from LDS-staged tile: lane l holds column n=l%16;
// K = {base..base+7} in VGPR0-3, {16+base..16+base+7} in VGPR4-7, base = (l>=16)*8.
__device__ inline v16bf load_b_frag_lds(const unsigned short* tile, int nsub, int ks, int lane) {
    const unsigned short* p =
        tile + (nsub * 16 + (lane & 15)) * LDSW + ks * 32 + ((lane & 16) ? 8 : 0);
    v8bf lo = *(const v8bf*)p;
    v8bf hi = *(const v8bf*)(p + 16);
    v16bf b;
    #pragma unroll
    for (int i = 0; i < 8; ++i) { b[i] = lo[i]; b[8 + i] = hi[i]; }
    return b;
}

// ---- stage a 32(N) x 128(K) bf16 weight tile into LDS (async on CDNA5) ----
__device__ inline void stage_copy(unsigned short* lds, const unsigned short* gbase,
                                  int k0, int tid) {
    #pragma unroll
    for (int j = 0; j < 4; ++j) {
        int id = tid + j * 128;          // 512 chunks of 16B, 128 threads
        int r = id >> 4, c = id & 15;
        const unsigned short* src = gbase + (size_t)r * KP + k0 + c * 8;
        unsigned short* dst = lds + r * LDSW + c * 8;
#if USE_ASYNC
        __builtin_amdgcn_global_load_async_to_lds_b128(
            (__attribute__((address_space(1))) v4i*)src,
            (__attribute__((address_space(3))) v4i*)dst, 0, 0);
#else
        *(uint4*)dst = *(const uint4*)src;
#endif
    }
}

// ---------------- fused layer-1 GEMM: h[B][640] = relu(feats @ [W1_actor | W1_critic*]) ----------------
// grid (B/128, NOUT/32), block (32,4): each wave owns a 32Mx32N tile;
// the 4 waves of a block share the LDS-staged, double-buffered B tile.
__global__ __launch_bounds__(128)
void fused_gemm(const unsigned short* __restrict__ feats,
                const unsigned short* __restrict__ wt,    // [NOUT][KP] bf16
                const float* __restrict__ ab1,            // [HA]
                const float* __restrict__ cb1,            // [T][HC]
                float* __restrict__ h) {                  // [B][NOUT]
    __shared__ unsigned short smem[2][32 * LDSW];
    const int lane = threadIdx.x;
    const int wv = threadIdx.y;
    const int tid = wv * 32 + lane;
    const int m0 = blockIdx.x * 128 + wv * 32;
    const int n0 = blockIdx.y * 32;

    const unsigned short* arow0 = feats + (size_t)(m0 + (lane & 15)) * KP;
    const unsigned short* arow1 = arow0 + (size_t)16 * KP;
    const unsigned short* gb = wt + (size_t)n0 * KP;

    v8f acc00 = {}, acc01 = {}, acc10 = {}, acc11 = {};

    stage_copy(smem[0], gb, 0, tid);
#if USE_ASYNC
    __builtin_amdgcn_s_wait_asynccnt(0);
#endif
    __syncthreads();

    for (int s = 0; s < NS; ++s) {
        if (s + 1 < NS) stage_copy(smem[(s + 1) & 1], gb, (s + 1) * 128, tid);
        const unsigned short* bt = smem[s & 1];
        #pragma unroll
        for (int ks = 0; ks < 4; ++ks) {
            const int k0 = s * 128 + ks * 32;
            v16bf va0 = load_a_frag(arow0, k0, lane);
            v16bf va1 = load_a_frag(arow1, k0, lane);
            v16bf vb0 = load_b_frag_lds(bt, 0, ks, lane);
            v16bf vb1 = load_b_frag_lds(bt, 1, ks, lane);
            acc00 = __builtin_amdgcn_wmma_f32_16x16x32_bf16(false, va0, false, vb0,
                                                            (short)0, acc00, false, false);
            acc01 = __builtin_amdgcn_wmma_f32_16x16x32_bf16(false, va0, false, vb1,
                                                            (short)0, acc01, false, false);
            acc10 = __builtin_amdgcn_wmma_f32_16x16x32_bf16(false, va1, false, vb0,
                                                            (short)0, acc10, false, false);
            acc11 = __builtin_amdgcn_wmma_f32_16x16x32_bf16(false, va1, false, vb1,
                                                            (short)0, acc11, false, false);
        }
#if USE_ASYNC
        __builtin_amdgcn_s_wait_asynccnt(0);   // stage s+1 copy is complete
#endif
        __syncthreads();
    }

    // C layout: lane l, VGPR r -> M = mtile + r + 8*(l/16), N = ntile + l%16
    const int nlo = n0 + (lane & 15);
    const int nhi = nlo + 16;
    const int mb = m0 + ((lane >> 4) << 3);
    const float b_lo = (nlo < HA) ? ab1[nlo] : cb1[nlo - HA];
    const float b_hi = (nhi < HA) ? ab1[nhi] : cb1[nhi - HA];
    #pragma unroll
    for (int r = 0; r < 8; ++r) {
        h[(size_t)(mb + r) * NOUT + nlo]      = fmaxf(acc00[r] + b_lo, 0.0f);
        h[(size_t)(mb + r) * NOUT + nhi]      = fmaxf(acc01[r] + b_hi, 0.0f);
        h[(size_t)(mb + 16 + r) * NOUT + nlo] = fmaxf(acc10[r] + b_lo, 0.0f);
        h[(size_t)(mb + 16 + r) * NOUT + nhi] = fmaxf(acc11[r] + b_hi, 0.0f);
    }
}

// ---------------- heads: log-softmax / chosen lp / neg-entropy / task critic ----------------
__global__ __launch_bounds__(256)
void head_kernel(const float* __restrict__ hAll, const float* __restrict__ w2,
                 const float* __restrict__ b2, const float* __restrict__ cw2,
                 const float* __restrict__ cb2, const int* __restrict__ actions,
                 const int* __restrict__ tasks, float* __restrict__ out) {
    int b = blockIdx.x * blockDim.x + threadIdx.x;
    if (b >= B) return;
    const float* ha = hAll + (size_t)b * NOUT;
    float s[A];
    #pragma unroll
    for (int a = 0; a < A; ++a) s[a] = b2[a];
    for (int hh = 0; hh < HA; ++hh) {
        float v = ha[hh];
        #pragma unroll
        for (int a = 0; a < A; ++a) s[a] += v * w2[hh * A + a];
    }
    float m = s[0];
    #pragma unroll
    for (int a = 1; a < A; ++a) m = fmaxf(m, s[a]);
    float sum = 0.0f;
    #pragma unroll
    for (int a = 0; a < A; ++a) sum += __expf(s[a] - m);
    float lse = __logf(sum) + m;
    float lp = s[actions[b]] - lse;
    float ne = 0.0f;
    #pragma unroll
    for (int a = 0; a < A; ++a) { float l = s[a] - lse; ne += __expf(l) * l; }
    int t = tasks[b];
    const float* hc = ha + HA + t * HC;
    const float* w = cw2 + t * HC;
    float cs = cb2[t];
    #pragma unroll
    for (int hh = 0; hh < HC; ++hh) cs += hc[hh] * w[hh];
    out[b * 3 + 0] = lp;
    out[b * 3 + 1] = cs;
    out[b * 3 + 2] = ne;
}

// ---------------- launcher ----------------
extern "C" void kernel_launch(void* const* d_in, const int* in_sizes, int n_in,
                              void* d_out, int out_size, void* d_ws, size_t ws_size,
                              hipStream_t stream) {
    const float* map_inputs  = (const float*)d_in[0];
    const float* flat_inputs = (const float*)d_in[1];
    const int*   task_ids    = (const int*)d_in[2];
    const int*   actions     = (const int*)d_in[3];
    const float* conv_w      = (const float*)d_in[4];
    const float* conv_b      = (const float*)d_in[5];
    const float* actor_w1    = (const float*)d_in[6];
    const float* actor_b1    = (const float*)d_in[7];
    const float* actor_w2    = (const float*)d_in[8];
    const float* actor_b2    = (const float*)d_in[9];
    const float* critic_w1   = (const float*)d_in[10];
    const float* critic_b1   = (const float*)d_in[11];
    const float* critic_w2   = (const float*)d_in[12];
    const float* critic_b2   = (const float*)d_in[13];
    float* out = (float*)d_out;

    // workspace carve-out (256B aligned)
    char* ws = (char*)d_ws;
    size_t off = 0;
    auto carve = [&](size_t bytes) -> char* {
        char* p = ws + off;
        off = (off + bytes + 255) & ~(size_t)255;
        return p;
    };
    unsigned short* feats = (unsigned short*)carve((size_t)B * KP * 2);     // ~59 MB
    unsigned short* wtAll = (unsigned short*)carve((size_t)NOUT * KP * 2);  // ~37 MB
    float*          hAll  = (float*)carve((size_t)B * NOUT * 4);            // ~2.6 MB
    (void)ws_size; (void)in_sizes; (void)n_in; (void)out_size;

    // 1) fused weight conversion (fp32 -> transposed, K-padded bf16)
    {
        size_t total = (size_t)NOUT * KP;
        cvt_weights<<<(unsigned)((total + 255) / 256), 256, 0, stream>>>(actor_w1,
                                                                         critic_w1, wtAll);
    }

    // 2) conv + featurize -> bf16 feats
    conv_feat<<<dim3(B, CONV), 128, 0, stream>>>(map_inputs, conv_w, conv_b,
                                                 flat_inputs, feats);

    // 3) fused actor+critic layer-1 WMMA GEMM: [1024 x 28928] x [28928 x 640]
    fused_gemm<<<dim3(B / 128, NOUT / 32), dim3(32, 4), 0, stream>>>(feats, wtAll,
                                                                     actor_b1, critic_b1,
                                                                     hAll);

    // 4) heads -> [B,3]
    head_kernel<<<(B + 255) / 256, 256, 0, stream>>>(hAll, actor_w2, actor_b2,
                                                     critic_w2, critic_b2,
                                                     actions, task_ids, out);
}
